// PerceiverNM_1494648619016
// MI455X (gfx1250) — matrix operand order repaired and checked by reading, not compiled, not run on hardware
//
#include <hip/hip_runtime.h>

// ---------------------------------------------------------------------------
// POYO / Perceiver forward for MI455X (gfx1250, wave32, WMMA).
// All dense matmuls run through one batched f16 WMMA GEMM (f32 accumulate),
// 128x128x64 tiles, double-buffered LDS fed by gfx1250 async global->LDS
// copies (ASYNCcnt), fused bias+residual epilogue.
// ---------------------------------------------------------------------------

typedef _Float16 h16;
typedef __attribute__((ext_vector_type(16))) _Float16 v16h;
typedef __attribute__((ext_vector_type(8)))  float    v8f;
typedef __attribute__((__vector_size__(16))) int      v4i;

#define TBM 128
#define TBN 128
#define TBK 64

#if __has_builtin(__builtin_amdgcn_global_load_async_to_lds_b128)
#define HAVE_ASYNC_LDS 1
#else
#define HAVE_ASYNC_LDS 0
#endif

__device__ __forceinline__ void async_copy16(const void* g, void* l)
{
#if HAVE_ASYNC_LDS
  __builtin_amdgcn_global_load_async_to_lds_b128((v4i*)g, (v4i*)l, 0, 0);
#else
  (void)g; (void)l;
#endif
}

__device__ __forceinline__ void async_wait0()
{
#if HAVE_ASYNC_LDS
#if __has_builtin(__builtin_amdgcn_s_wait_asynccnt)
  __builtin_amdgcn_s_wait_asynccnt(0);
#else
  asm volatile("s_wait_asynccnt 0" ::: "memory");
#endif
#endif
}

// ============================ WMMA GEMM ====================================
// C[M][N](f32) = A[M][K](f16,row-major) * Bt[N][K](f16 = B transposed)
//                (+ bias[n]) (+ Res[m][n])     batched via blockIdx.z
__global__ __launch_bounds__(256, 1) void k_gemm(
    const h16* __restrict__ A,  long long sAb, int lda,
    const h16* __restrict__ Bt, long long sBb, int ldb,
    float* __restrict__ C,      long long sCb, int ldc,
    const float* __restrict__ bias,
    const float* __restrict__ Res, long long sRb, int ldr,
    int M, int N, int K)
{
  __shared__ h16 shA[2][TBM][TBK];   // 32 KB
  __shared__ h16 shB[2][TBN][TBK];   // 32 KB

  const int zb = blockIdx.z;
  A  += (long long)zb * sAb;
  Bt += (long long)zb * sBb;
  C  += (long long)zb * sCb;
  if (Res) Res += (long long)zb * sRb;

  const int m0 = blockIdx.y * TBM;
  const int n0 = blockIdx.x * TBN;

  const int t    = threadIdx.x;
  const int lane = t & 31;
  const int w    = t >> 5;
  const int wr   = w & 3;    // 4 row-groups of 32 rows
  const int wc   = w >> 2;   // 2 col-groups of 64 cols
  const int l15  = lane & 15;
  const int g    = lane >> 4;

  // global->LDS staging map: 2 threads per tile-row, 32 halfs (64B) each
  const int lr = t >> 1;
  const int lc = (t & 1) * 32;
  int ar = m0 + lr; if (ar > M - 1) ar = M - 1;
  int br = n0 + lr; if (br > N - 1) br = N - 1;
  const h16* gA = A  + (long long)ar * lda + lc;
  const h16* gB = Bt + (long long)br * ldb + lc;

  v8f acc[2][4];
  const v8f vzero = {0.f,0.f,0.f,0.f,0.f,0.f,0.f,0.f};
#pragma unroll
  for (int i = 0; i < 2; ++i)
#pragma unroll
    for (int j = 0; j < 4; ++j) acc[i][j] = vzero;

  const int nk = K / TBK;

  // prologue: stage tile 0
#if HAVE_ASYNC_LDS
#pragma unroll
  for (int i = 0; i < 4; ++i) {
    async_copy16(gA + i * 8, &shA[0][lr][lc + i * 8]);
    async_copy16(gB + i * 8, &shB[0][lr][lc + i * 8]);
  }
  async_wait0();
#else
#pragma unroll
  for (int i = 0; i < 4; ++i) {
    *(uint4*)&shA[0][lr][lc + i * 8] = ((const uint4*)gA)[i];
    *(uint4*)&shB[0][lr][lc + i * 8] = ((const uint4*)gB)[i];
  }
#endif
  __syncthreads();

  for (int kt = 0; kt < nk; ++kt) {
    const int buf = kt & 1;
    const bool haveNext = (kt + 1) < nk;

#if HAVE_ASYNC_LDS
    if (haveNext) {
      // issue async copies for the next tile; no VGPR staging at all
      const int kof = (kt + 1) * TBK;
      const int nb = buf ^ 1;
#pragma unroll
      for (int i = 0; i < 4; ++i) {
        async_copy16(gA + kof + i * 8, &shA[nb][lr][lc + i * 8]);
        async_copy16(gB + kof + i * 8, &shB[nb][lr][lc + i * 8]);
      }
      if (kt + 2 < nk) {   // hint L2: tile after next
        __builtin_prefetch((const void*)(gA + (kt + 2) * TBK), 0, 1);
        __builtin_prefetch((const void*)(gB + (kt + 2) * TBK), 0, 1);
      }
    }
#else
    uint4 ra[4], rb[4];
    if (haveNext) {
      const int kof = (kt + 1) * TBK;
      const uint4* pa = (const uint4*)(gA + kof);
      const uint4* pb = (const uint4*)(gB + kof);
#pragma unroll
      for (int i = 0; i < 4; ++i) { ra[i] = pa[i]; rb[i] = pb[i]; }
      if (kt + 2 < nk) {
        __builtin_prefetch((const void*)(gA + (kt + 2) * TBK), 0, 1);
        __builtin_prefetch((const void*)(gB + (kt + 2) * TBK), 0, 1);
      }
    }
#endif

#pragma unroll
    for (int kk = 0; kk < TBK; kk += 32) {
      v16h af[2], bf[4];
#pragma unroll
      for (int i = 0; i < 2; ++i) {
        // A 16x32 f16 fragment: lane holds row (l15), K runs {8g..}, {16+8g..}
        const h16* p = &shA[buf][wr * 32 + i * 16 + l15][kk];
        ((uint4*)&af[i])[0] = *(const uint4*)(p + 8 * g);
        ((uint4*)&af[i])[1] = *(const uint4*)(p + 16 + 8 * g);
      }
#pragma unroll
      for (int j = 0; j < 4; ++j) {
        // B 32x16 fragment from Bt tile: lane holds col (l15), K runs {8g..}, {16+8g..}
        const h16* p = &shB[buf][wc * 64 + j * 16 + l15][kk];
        ((uint4*)&bf[j])[0] = *(const uint4*)(p + 8 * g);
        ((uint4*)&bf[j])[1] = *(const uint4*)(p + 16 + 8 * g);
      }
#pragma unroll
      for (int i = 0; i < 2; ++i)
#pragma unroll
        for (int j = 0; j < 4; ++j)
          acc[i][j] = __builtin_amdgcn_wmma_f32_16x16x32_f16(
              false, af[i], false, bf[j], (short)0, acc[i][j], false, false);
    }

#if HAVE_ASYNC_LDS
    if (haveNext) async_wait0();   // own wave's async copies done
#else
    if (haveNext) {
      const int nb = buf ^ 1;
#pragma unroll
      for (int i = 0; i < 4; ++i) {
        *(uint4*)&shA[nb][lr][lc + i * 8] = ra[i];
        *(uint4*)&shB[nb][lr][lc + i * 8] = rb[i];
      }
    }
#endif
    __syncthreads();
  }

  // epilogue: D layout: lane(0..15)=n, vgpr r -> m = 8*(lane>>4)+r
#pragma unroll
  for (int j = 0; j < 4; ++j) {
    const int gn = n0 + wc * 64 + j * 16 + l15;
    if (gn >= N) continue;
    const float bv = bias ? bias[gn] : 0.0f;
#pragma unroll
    for (int i = 0; i < 2; ++i) {
#pragma unroll
      for (int r = 0; r < 8; ++r) {
        const int gm = m0 + wr * 32 + i * 16 + g * 8 + r;
        if (gm >= M) continue;
        float v = acc[i][j][r] + bv;
        if (Res) v += Res[(long long)gm * ldr + gn];
        C[(long long)gm * ldc + gn] = v;
      }
    }
  }
}

// ===================== weight convert + transpose ==========================
// W[K][N] f32  ->  Wt[N][K] f16
__global__ void k_cvt_wT(const float* __restrict__ W, h16* __restrict__ Wt,
                         int K, int N)
{
  long long i = (long long)blockIdx.x * blockDim.x + threadIdx.x;
  long long total = (long long)K * N;
  if (i >= total) return;
  int n = (int)(i / K), k = (int)(i % K);
  Wt[i] = (h16)W[(long long)k * N + n];
}

// ========================= gather (embedding) ==============================
__global__ void k_gather(const int* __restrict__ idx, const float* __restrict__ emb,
                         float* __restrict__ out, long long total)
{
  long long i = (long long)blockIdx.x * blockDim.x + threadIdx.x;
  if (i >= total) return;
  long long r = i >> 9; int c = (int)(i & 511);
  out[i] = emb[(long long)idx[r] * 512 + c];
}

// ================== fused embed(add) + layernorm -> f16 ====================
__global__ __launch_bounds__(128) void k_embed_ln(
    const int* __restrict__ uid, const int* __restrict__ tix,
    const float* __restrict__ Eu, const float* __restrict__ Et,
    const float* __restrict__ gma, const float* __restrict__ bta,
    h16* __restrict__ out)
{
  __shared__ float red[128];
  const long long row = blockIdx.x;
  const int t = threadIdx.x;
  const float* eu = Eu + (long long)uid[row] * 512;
  const float* et = Et + (long long)tix[row] * 512;
  float x[4]; float s = 0.f;
#pragma unroll
  for (int i = 0; i < 4; ++i) { x[i] = eu[t * 4 + i] + et[t * 4 + i]; s += x[i]; }
  red[t] = s; __syncthreads();
  for (int k = 64; k > 0; k >>= 1) { if (t < k) red[t] += red[t + k]; __syncthreads(); }
  const float mu = red[0] * (1.f / 512.f); __syncthreads();
  float vs = 0.f;
#pragma unroll
  for (int i = 0; i < 4; ++i) { float d = x[i] - mu; vs += d * d; }
  red[t] = vs; __syncthreads();
  for (int k = 64; k > 0; k >>= 1) { if (t < k) red[t] += red[t + k]; __syncthreads(); }
  const float rs = rsqrtf(red[0] * (1.f / 512.f) + 1e-5f); __syncthreads();
#pragma unroll
  for (int i = 0; i < 4; ++i) {
    int c = t * 4 + i;
    out[row * 512 + c] = (h16)((x[i] - mu) * rs * gma[c] + bta[c]);
  }
}

// ===================== layernorm(f32 row) -> f16 ===========================
__global__ __launch_bounds__(128) void k_ln16(
    const float* __restrict__ in, const float* __restrict__ gma,
    const float* __restrict__ bta, h16* __restrict__ out)
{
  __shared__ float red[128];
  const long long row = blockIdx.x;
  const int t = threadIdx.x;
  float x[4]; float s = 0.f;
#pragma unroll
  for (int i = 0; i < 4; ++i) { x[i] = in[row * 512 + t * 4 + i]; s += x[i]; }
  red[t] = s; __syncthreads();
  for (int k = 64; k > 0; k >>= 1) { if (t < k) red[t] += red[t + k]; __syncthreads(); }
  const float mu = red[0] * (1.f / 512.f); __syncthreads();
  float vs = 0.f;
#pragma unroll
  for (int i = 0; i < 4; ++i) { float d = x[i] - mu; vs += d * d; }
  red[t] = vs; __syncthreads();
  for (int k = 64; k > 0; k >>= 1) { if (t < k) red[t] += red[t + k]; __syncthreads(); }
  const float rs = rsqrtf(red[0] * (1.f / 512.f) + 1e-5f); __syncthreads();
#pragma unroll
  for (int i = 0; i < 4; ++i) {
    int c = t * 4 + i;
    out[row * 512 + c] = (h16)((x[i] - mu) * rs * gma[c] + bta[c]);
  }
}

// ================ RoPE / convert, with layout remap to f16 =================
// in_mode : 0 merged  in[(b*N+n)*irs + ico + h*64 + d]
//           1 BHND    in[((b*H+h)*N+n)*64 + d]
// out_mode: 0 merged  out[(b*N+n)*ors + oco + h*64 + d]
//           1 BHND    out[((b*H+h)*N+n)*64 + d]
//           2 BH_DN   out[((b*H+h)*64+d)*N + n]      (pre-transposed V)
__global__ void k_rope(const float* __restrict__ in, int in_mode, long long irs, int ico,
                       h16* __restrict__ out, int out_mode, long long ors, int oco,
                       const float* __restrict__ ts,
                       int H, int N, float sign, int do_rope, long long total)
{
  long long idx = (long long)blockIdx.x * blockDim.x + threadIdx.x;
  if (idx >= total) return;
  const int j = (int)(idx & 31);          // rotation pair 0..31
  long long r = idx >> 5;                 // (b*H+h)*N + n
  const int n = (int)(r % N);
  const long long bh = r / N;
  const int h = (int)(bh % H);
  const long long b = bh / H;

  const float* src = (in_mode == 0)
      ? in + ((b * N + n) * irs + ico + h * 64 + 2 * j)
      : in + ((bh * N + n) * 64 + 2 * j);
  float x0 = src[0], x1 = src[1];
  float y0 = x0, y1 = x1;
  if (do_rope) {
    float inv = 0.0f;
    if (j < 16) inv = 6.28318530718f / (1e-4f * powf(4.0e4f, (float)j * (1.0f / 16.0f)));
    const float a = ts[b * N + n] * inv * sign;
    const float c = cosf(a), s = sinf(a);
    y0 = x0 * c - x1 * s;
    y1 = x1 * c + x0 * s;
  }
  if (out_mode == 2) {
    const int d = 2 * j;
    out[(bh * 64 + d)     * (long long)N + n] = (h16)y0;
    out[(bh * 64 + d + 1) * (long long)N + n] = (h16)y1;
  } else {
    h16* dst = (out_mode == 0)
        ? out + ((b * N + n) * ors + oco + h * 64 + 2 * j)
        : out + ((bh * N + n) * 64 + 2 * j);
    dst[0] = (h16)y0; dst[1] = (h16)y1;
  }
}

// ============================ softmax ======================================
__global__ __launch_bounds__(256) void k_softmax(
    const float* __restrict__ S, h16* __restrict__ P,
    const unsigned char* __restrict__ mask, int rows_per_b, int L, float scale)
{
  __shared__ float red[256];
  const long long row = blockIdx.x;
  const long long base = row * (long long)L;
  const unsigned char* mrow = mask ? mask + (long long)(row / rows_per_b) * L : nullptr;
  const int t = threadIdx.x;
  float mx = -3.0e38f;
  for (int c = t; c < L; c += 256) {
    float v = S[base + c] * scale;
    if (mrow && !mrow[c]) v = -1.0e30f;
    mx = fmaxf(mx, v);
  }
  red[t] = mx; __syncthreads();
  for (int k = 128; k > 0; k >>= 1) { if (t < k) red[t] = fmaxf(red[t], red[t + k]); __syncthreads(); }
  mx = red[0]; __syncthreads();
  float sum = 0.f;
  for (int c = t; c < L; c += 256) {
    float v = S[base + c] * scale;
    if (mrow && !mrow[c]) v = -1.0e30f;
    float e = expf(v - mx);
    sum += e;
    P[base + c] = (h16)e;
  }
  red[t] = sum; __syncthreads();
  for (int k = 128; k > 0; k >>= 1) { if (t < k) red[t] += red[t + k]; __syncthreads(); }
  const float invs = 1.0f / red[0]; __syncthreads();
  for (int c = t; c < L; c += 256)
    P[base + c] = (h16)((float)P[base + c] * invs);
}

// ===================== GEGLU: a * gelu_exact(g) -> f16 =====================
__global__ void k_geglu(const float* __restrict__ h, h16* __restrict__ out, long long total)
{
  long long i = (long long)blockIdx.x * blockDim.x + threadIdx.x;
  if (i >= total) return;
  long long r = i >> 11; int c = (int)(i & 2047);
  float a = h[r * 4096 + c];
  float g = h[r * 4096 + 2048 + c];
  float ge = 0.5f * g * (1.0f + erff(g * 0.70710678118f));
  out[i] = (h16)(a * ge);
}

// =============== final: LN(row) . Wout + bout  (ODIM = 1) ==================
__global__ __launch_bounds__(128) void k_out(
    const float* __restrict__ q, const float* __restrict__ gma,
    const float* __restrict__ bta, const float* __restrict__ Wout,
    const float* __restrict__ bout, float* __restrict__ out)
{
  __shared__ float red[128];
  const long long row = blockIdx.x;
  const int t = threadIdx.x;
  float x[4]; float s = 0.f;
#pragma unroll
  for (int i = 0; i < 4; ++i) { x[i] = q[row * 512 + t * 4 + i]; s += x[i]; }
  red[t] = s; __syncthreads();
  for (int k = 64; k > 0; k >>= 1) { if (t < k) red[t] += red[t + k]; __syncthreads(); }
  const float mu = red[0] * (1.f / 512.f); __syncthreads();
  float vs = 0.f;
#pragma unroll
  for (int i = 0; i < 4; ++i) { float d = x[i] - mu; vs += d * d; }
  red[t] = vs; __syncthreads();
  for (int k = 64; k > 0; k >>= 1) { if (t < k) red[t] += red[t + k]; __syncthreads(); }
  const float rs = rsqrtf(red[0] * (1.f / 512.f) + 1e-5f); __syncthreads();
  float dot = 0.f;
#pragma unroll
  for (int i = 0; i < 4; ++i) {
    int c = t * 4 + i;
    dot += ((x[i] - mu) * rs * gma[c] + bta[c]) * Wout[c];
  }
  red[t] = dot; __syncthreads();
  for (int k = 64; k > 0; k >>= 1) { if (t < k) red[t] += red[t + k]; __syncthreads(); }
  if (t == 0) out[row] = red[0] + bout[0];
}

// =========================== host orchestration ============================
static void gemm(hipStream_t st, const h16* A, long long sA, int lda,
                 const h16* Bt, long long sB, int ldb,
                 float* C, long long sC, int ldc,
                 const float* bias, const float* Res, long long sR, int ldr,
                 int M, int N, int K, int batches)
{
  dim3 g((N + TBN - 1) / TBN, (M + TBM - 1) / TBM, batches);
  k_gemm<<<g, 256, 0, st>>>(A, sA, lda, Bt, sB, ldb, C, sC, ldc,
                            bias, Res, sR, ldr, M, N, K);
}

static void rope(hipStream_t st, const float* in, int im, long long irs, int ico,
                 h16* out, int om, long long ors, int oco,
                 const float* ts, int H, int N, float sign, int dorope)
{
  long long total = 16LL * H * N * 32;
  k_rope<<<(unsigned)((total + 255) / 256), 256, 0, st>>>(
      in, im, irs, ico, out, om, ors, oco, ts, H, N, sign, dorope, total);
}

static void cvtw(hipStream_t st, const float* W, h16* Wt, int K, int N)
{
  long long total = (long long)K * N;
  k_cvt_wT<<<(unsigned)((total + 255) / 256), 256, 0, st>>>(W, Wt, K, N);
}

extern "C" void kernel_launch(void* const* d_in, const int* in_sizes, int n_in,
                              void* d_out, int out_size, void* d_ws, size_t ws_size,
                              hipStream_t stream)
{
  (void)in_sizes; (void)n_in; (void)out_size; (void)ws_size;
  const int B = 16, NIN = 8192, M = 512, NOUT = 1024, DIM = 512;

  // ---- inputs (setup_inputs order) ----
  const int*   suid  = (const int*)  d_in[0];
  const int*   stidx = (const int*)  d_in[1];
  const float* sts   = (const float*)d_in[2];
  const unsigned char* imask = (const unsigned char*)d_in[3];
  const int*   latix = (const int*)  d_in[4];
  const float* lts   = (const float*)d_in[5];
  const int*   sessi = (const int*)  d_in[6];
  const float* ots   = (const float*)d_in[7];
  // ---- params: jax sorted-dict flatten order ----
#define PRM(j) ((const float*)d_in[8 + (j)])
  const float *Wout = PRM(0), *bout = PRM(1);
  const float *dWkv = PRM(2), *dWo = PRM(3), *dWq = PRM(4), *dbo = PRM(5);
  const float *dlncb = PRM(6), *dlncg = PRM(7), *dlnqb = PRM(8), *dlnqg = PRM(9);
  const float *dec_ln_b = PRM(10), *dec_ln_g = PRM(11);
  const float *eWkv = PRM(12), *eWo = PRM(13), *eWq = PRM(14), *ebo = PRM(15);
  const float *elncb = PRM(16), *elncg = PRM(17), *elnqb = PRM(18), *elnqg = PRM(19);
  const float *fW1 = PRM(20), *fW2 = PRM(21), *fb1 = PRM(22), *fb2 = PRM(23);
  const float *flnb = PRM(24), *flng = PRM(25);
  const float *latent_emb = PRM(26);
  const float *session_emb = PRM(49), *type_emb = PRM(50), *unit_emb = PRM(51);

  // ---- workspace carve ----
  char* wp = (char*)d_ws;
  auto alloc = [&](size_t bytes) -> void* {
    void* r = wp; wp += (bytes + 255) & ~(size_t)255; return r;
  };
  h16* encWqT  = (h16*)alloc((size_t)64 * 512 * 2);
  h16* encWkvT = (h16*)alloc((size_t)128 * 512 * 2);
  h16* encWoT  = (h16*)alloc((size_t)512 * 64 * 2);
  h16* effW1T  = (h16*)alloc((size_t)4096 * 512 * 2);
  h16* effW2T  = (h16*)alloc((size_t)512 * 2048 * 2);
  h16 *pWqkvT[2], *pWoT[2], *pW1T[2], *pW2T[2];
  for (int l = 0; l < 2; ++l) {
    pWqkvT[l] = (h16*)alloc((size_t)1536 * 512 * 2);
    pWoT[l]   = (h16*)alloc((size_t)512 * 512 * 2);
    pW1T[l]   = (h16*)alloc((size_t)4096 * 512 * 2);
    pW2T[l]   = (h16*)alloc((size_t)512 * 2048 * 2);
  }
  h16* decWqT  = (h16*)alloc((size_t)64 * 512 * 2);
  h16* decWkvT = (h16*)alloc((size_t)128 * 512 * 2);
  h16* decWoT  = (h16*)alloc((size_t)512 * 64 * 2);

  h16*   xc16 = (h16*)  alloc((size_t)B * NIN * DIM * 2);     // 128 MB
  float* latA = (float*)alloc((size_t)B * M * DIM * 4);
  float* latB = (float*)alloc((size_t)B * M * DIM * 4);
  h16*   n16a = (h16*)  alloc((size_t)B * M * DIM * 2);
  h16*   n16b = (h16*)  alloc((size_t)B * NOUT * DIM * 2);
  float* bigA = (float*)alloc((size_t)B * M * NIN * 4);       // 256 MB (S / h / kv / qkv)
  float* bigB = (float*)alloc((size_t)128 * 512 * 64 * 4);    // attn O / q proj scratch
  h16*   P16  = (h16*)  alloc((size_t)B * M * NIN * 2);       // 128 MB
  h16*   q16  = (h16*)  alloc((size_t)128 * 512 * 64 * 2);
  h16*   k16  = (h16*)  alloc((size_t)B * NIN * 64 * 2);
  h16*   vT16 = (h16*)  alloc((size_t)B * NIN * 64 * 2);
  h16*   o16  = (h16*)  alloc((size_t)B * M * DIM * 2);
  h16*   ag16 = (h16*)  alloc((size_t)B * M * 2048 * 2);
  float* qdA  = (float*)alloc((size_t)B * NOUT * DIM * 4);
  float* qdB  = (float*)alloc((size_t)B * NOUT * DIM * 4);

  // ---- weights -> f16 transposed ----
  cvtw(stream, eWq, encWqT, 512, 64);
  cvtw(stream, eWkv, encWkvT, 512, 128);
  cvtw(stream, eWo, encWoT, 64, 512);
  cvtw(stream, fW1, effW1T, 512, 4096);
  cvtw(stream, fW2, effW2T, 2048, 512);
  for (int l = 0; l < 2; ++l) {
    int j = 27 + l * 11;
    cvtw(stream, PRM(j + 1), pWqkvT[l], 512, 1536);  // attn.Wqkv
    cvtw(stream, PRM(j + 0), pWoT[l],   512, 512);   // attn.Wo
    cvtw(stream, PRM(j + 5), pW1T[l],   512, 4096);  // ffn.W1
    cvtw(stream, PRM(j + 6), pW2T[l],   2048, 512);  // ffn.W2
  }
  cvtw(stream, dWq, decWqT, 512, 64);
  cvtw(stream, dWkv, decWkvT, 512, 128);
  cvtw(stream, dWo, decWoT, 64, 512);

  // ---- embeddings ----
  k_embed_ln<<<B * NIN, 128, 0, stream>>>(suid, stidx, unit_emb, type_emb,
                                          elncg, elncb, xc16);
  k_gather<<<(unsigned)(((long long)B * M * DIM + 255) / 256), 256, 0, stream>>>(
      latix, latent_emb, latA, (long long)B * M * DIM);

  // ============== encoder cross-attn (rotate_value=True) ==================
  k_ln16<<<B * M, 128, 0, stream>>>(latA, elnqg, elnqb, n16a);
  gemm(stream, n16a, 0, 512, encWqT, 0, 512, bigB, 0, 64,
       nullptr, nullptr, 0, 0, B * M, 64, 512, 1);
  rope(stream, bigB, 0, 64, 0, q16, 1, 0, 0, lts, 1, M, +1.f, 1);
  gemm(stream, xc16, 0, 512, encWkvT, 0, 512, bigA, 0, 128,
       nullptr, nullptr, 0, 0, B * NIN, 128, 512, 1);
  rope(stream, bigA, 0, 128, 0,  k16,  1, 0, 0, sts, 1, NIN, +1.f, 1);
  rope(stream, bigA, 0, 128, 64, vT16, 2, 0, 0, sts, 1, NIN, +1.f, 1);
  gemm(stream, q16, (long long)M * 64, 64, k16, (long long)NIN * 64, 64,
       bigA, (long long)M * NIN, NIN, nullptr, nullptr, 0, 0, M, NIN, 64, B);
  k_softmax<<<B * M, 256, 0, stream>>>(bigA, P16, imask, M, NIN, 0.125f);
  gemm(stream, P16, (long long)M * NIN, NIN, vT16, (long long)64 * NIN, NIN,
       bigB, (long long)M * 64, 64, nullptr, nullptr, 0, 0, M, 64, NIN, B);
  rope(stream, bigB, 1, 0, 0, o16, 0, 64, 0, lts, 1, M, -1.f, 1);  // un-rotate
  gemm(stream, o16, 0, 64, encWoT, 0, 64, latB, 0, 512,
       ebo, latA, 0, 512, B * M, 512, 64, 1);

  // encoder FFN: latB -> latA
  k_ln16<<<B * M, 128, 0, stream>>>(latB, flng, flnb, n16a);
  gemm(stream, n16a, 0, 512, effW1T, 0, 512, bigA, 0, 4096,
       fb1, nullptr, 0, 0, B * M, 4096, 512, 1);
  k_geglu<<<(unsigned)(((long long)B * M * 2048 + 255) / 256), 256, 0, stream>>>(
      bigA, ag16, (long long)B * M * 2048);
  gemm(stream, ag16, 0, 2048, effW2T, 0, 2048, latA, 0, 512,
       fb2, latB, 0, 512, B * M, 512, 2048, 1);

  // ==================== processor self-attn layers ========================
  float* cur = latA; float* nxt = latB;
  for (int l = 0; l < 2; ++l) {
    int j = 27 + l * 11;
    k_ln16<<<B * M, 128, 0, stream>>>(cur, PRM(j + 4), PRM(j + 3), n16a);
    gemm(stream, n16a, 0, 512, pWqkvT[l], 0, 512, bigA, 0, 1536,
         nullptr, nullptr, 0, 0, B * M, 1536, 512, 1);
    rope(stream, bigA, 0, 1536, 0,    q16,  1, 0, 0, lts, 8, M, +1.f, 1);
    rope(stream, bigA, 0, 1536, 512,  k16,  1, 0, 0, lts, 8, M, +1.f, 1);
    rope(stream, bigA, 0, 1536, 1024, vT16, 2, 0, 0, lts, 8, M, +1.f, 1);
    gemm(stream, q16, (long long)M * 64, 64, k16, (long long)M * 64, 64,
         bigA, (long long)M * M, M, nullptr, nullptr, 0, 0, M, M, 64, B * 8);
    k_softmax<<<B * 8 * M, 256, 0, stream>>>(bigA, P16, nullptr, M, M, 0.125f);
    gemm(stream, P16, (long long)M * M, M, vT16, (long long)64 * M, M,
         bigB, (long long)M * 64, 64, nullptr, nullptr, 0, 0, M, 64, M, B * 8);
    rope(stream, bigB, 1, 0, 0, o16, 0, 512, 0, lts, 8, M, -1.f, 1);
    gemm(stream, o16, 0, 512, pWoT[l], 0, 512, nxt, 0, 512,
         PRM(j + 2), cur, 0, 512, B * M, 512, 512, 1);
    // ffn: nxt -> cur
    k_ln16<<<B * M, 128, 0, stream>>>(nxt, PRM(j + 10), PRM(j + 9), n16a);
    gemm(stream, n16a, 0, 512, pW1T[l], 0, 512, bigA, 0, 4096,
         PRM(j + 7), nullptr, 0, 0, B * M, 4096, 512, 1);
    k_geglu<<<(unsigned)(((long long)B * M * 2048 + 255) / 256), 256, 0, stream>>>(
        bigA, ag16, (long long)B * M * 2048);
    gemm(stream, ag16, 0, 2048, pW2T[l], 0, 2048, cur, 0, 512,
         PRM(j + 8), nxt, 0, 512, B * M, 512, 2048, 1);
  }
  // final latents in `cur` (latA)

  // =============== decoder cross-attn (rotate_value=False) ================
  k_gather<<<(unsigned)(((long long)B * NOUT * DIM + 255) / 256), 256, 0, stream>>>(
      sessi, session_emb, qdA, (long long)B * NOUT * DIM);
  k_ln16<<<B * NOUT, 128, 0, stream>>>(qdA, dlnqg, dlnqb, n16b);
  k_ln16<<<B * M, 128, 0, stream>>>(cur, dlncg, dlncb, n16a);
  gemm(stream, n16b, 0, 512, decWqT, 0, 512, bigB, 0, 64,
       nullptr, nullptr, 0, 0, B * NOUT, 64, 512, 1);
  rope(stream, bigB, 0, 64, 0, q16, 1, 0, 0, ots, 1, NOUT, +1.f, 1);
  gemm(stream, n16a, 0, 512, decWkvT, 0, 512, bigA, 0, 128,
       nullptr, nullptr, 0, 0, B * M, 128, 512, 1);
  rope(stream, bigA, 0, 128, 0,  k16,  1, 0, 0, lts, 1, M, +1.f, 1);
  rope(stream, bigA, 0, 128, 64, vT16, 2, 0, 0, lts, 1, M, 0.f, 0);   // plain V
  gemm(stream, q16, (long long)NOUT * 64, 64, k16, (long long)M * 64, 64,
       bigA, (long long)NOUT * M, M, nullptr, nullptr, 0, 0, NOUT, M, 64, B);
  k_softmax<<<B * NOUT, 256, 0, stream>>>(bigA, P16, nullptr, NOUT, M, 0.125f);
  gemm(stream, P16, (long long)NOUT * M, M, vT16, (long long)64 * M, M,
       bigB, (long long)NOUT * 64, 64, nullptr, nullptr, 0, 0, NOUT, 64, M, B);
  rope(stream, bigB, 1, 0, 0, o16, 0, 64, 0, ots, 1, NOUT, 0.f, 0);   // convert only
  gemm(stream, o16, 0, 64, decWoT, 0, 64, qdB, 0, 512,
       dbo, qdA, 0, 512, B * NOUT, 512, 64, 1);

  // final head: LN . Wout + bout
  k_out<<<B * NOUT, 128, 0, stream>>>(qdB, dec_ln_g, dec_ln_b, Wout, bout,
                                      (float*)d_out);
#undef PRM
}
